// THST_CLSTM_Input_Layer_58445914964728
// MI455X (gfx1250) — matrix operand order repaired and checked by reading, not compiled, not run on hardware
//
#include <hip/hip_runtime.h>
#include <hip/hip_bf16.h>
#include <math.h>

// ---------------------------------------------------------------------------
// Bidirectional ConvLSTM2D, B=8 T=16 H=W=64 Cin=32 F=64.
// Implicit GEMM on v_wmma_f32_16x16x32_f16 (wave32, gfx1250), halo-padded
// activations, weights pre-packed in B-fragment layout, and an explicit
// 1-deep software pipeline (load k-block i+1 while WMMAs of i issue).
// Block = 32 pixels x 256 channels, 4 waves (wave == gate), 8 acc/wave.
// ---------------------------------------------------------------------------

typedef __attribute__((ext_vector_type(16))) _Float16 v16h;
typedef __attribute__((ext_vector_type(8)))  float    v8f;
typedef __attribute__((ext_vector_type(4)))  float    f32x4;

union Frag16 { v16h v; f32x4 q[2]; };

// ---- workspace layout (bytes) ----
static constexpr size_t OFF_XPAD = 0;         // 8*16*66*66*32 f16 = 35,684,352 B
static constexpr size_t OFF_WXF  = 35684352;  // 288*256 f16 packed = 147,456 B
static constexpr size_t OFF_WHF  = 35831808;  // 576*256 f16 packed = 294,912 B
static constexpr size_t OFF_WXB  = 36126720;
static constexpr size_t OFF_WHB  = 36274176;
static constexpr size_t OFF_C    = 36569088;  // 2,097,152 f32 = 8,388,608 B
static constexpr size_t OFF_H0   = 44957696;  // 8*66*66*64 f16 = 4,460,544 B
static constexpr size_t OFF_H1   = 49418240;  // 8*66*66*64 f16

// ---------------------------------------------------------------------------
// Pack HWIO weights [K=9*Cin, N=256] into per-WMMA B fragments (f32->f16).
// Packed element ((kb*16+nt)*32+lane)*16+j = W[kb*32+(lane>=16?8:0)+(j&7)+(j>=8?16:0)][nt*16+(lane&15)]
// ---------------------------------------------------------------------------
__global__ void pack_w_kernel(const float* __restrict__ w,
                              _Float16* __restrict__ wp, int K) {
  int total = K * 256;
  for (int i = blockIdx.x * blockDim.x + threadIdx.x; i < total;
       i += gridDim.x * blockDim.x) {
    int frag = i >> 9;
    int lane = (i >> 4) & 31;
    int j    = i & 15;
    int kb   = frag >> 4;
    int nt   = frag & 15;
    int k = kb * 32 + ((lane >> 4) << 3) + (j & 7) + ((j >> 3) << 4);
    int n = nt * 16 + (lane & 15);
    wp[i] = (_Float16)w[k * 256 + n];
  }
}

__global__ void zero_f16_kernel(_Float16* __restrict__ p, int n) {
  for (int i = blockIdx.x * blockDim.x + threadIdx.x; i < n;
       i += gridDim.x * blockDim.x)
    p[i] = (_Float16)0.0f;
}

__global__ void zero_f32_kernel(float* __restrict__ p, int n) {
  for (int i = blockIdx.x * blockDim.x + threadIdx.x; i < n;
       i += gridDim.x * blockDim.x)
    p[i] = 0.0f;
}

// scatter-convert x [B,T,64,64,32] f32 -> padded [B,T,66,66,32] f16 interior
__global__ void cvt_pad_kernel(const float* __restrict__ src,
                               _Float16* __restrict__ dst, int n) {
  for (int i = blockIdx.x * blockDim.x + threadIdx.x; i < n;
       i += gridDim.x * blockDim.x) {
    int ci = i & 31;
    int xx = (i >> 5) & 63;
    int yy = (i >> 11) & 63;
    int tt = (i >> 17) & 15;
    int bb = i >> 21;
    size_t d = (((size_t)(bb * 16 + tt) * 66 + (yy + 1)) * 66 + (xx + 1)) * 32 + ci;
    dst[d] = (_Float16)src[i];
  }
}

__device__ __forceinline__ float hsig(float z) {
  return fminf(fmaxf(0.2f * z + 0.5f, 0.0f), 1.0f);
}

// ---------------------------------------------------------------------------
// One fused ConvLSTM timestep over 32 pixels x 256 gate-channels per block.
//   z = b + conv3x3(x_t, Wx) + conv3x3(h_prev, Wh)  -> gates -> c,h
// ---------------------------------------------------------------------------
__global__ __launch_bounds__(128)
void convlstm_step_kernel(const _Float16* __restrict__ xpad,
                          const _Float16* __restrict__ wxp,  // 9*16 frags
                          const _Float16* __restrict__ whp,  // 18*16 frags
                          const float* __restrict__ bias,    // 256
                          const _Float16* __restrict__ hprev, // padded
                          _Float16* __restrict__ hnext,       // padded
                          float* __restrict__ cbuf,           // unpadded
                          float* __restrict__ out,
                          int t, int dir) {
  const int lane = threadIdx.x & 31;
  const int wv   = threadIdx.x >> 5;       // wave id == gate id
  const int pgBase = blockIdx.x * 32;      // 32 consecutive pixels, same row

  // this lane's A rows: pixel pg0 (tile 0) and pg0+16 (tile 1)
  const int m  = lane & 15;
  const int pg = pgBase + m;
  const int bb = pg >> 12;                 // H*W = 4096
  const int yy = (pg >> 6) & 63;
  const int xx = pg & 63;                  // tile1 pixel = xx+16, same b,y
  const int kOff = (lane >> 4) << 3;       // 0 or 8
  const int nl = lane & 15;

  const _Float16* xfrm = xpad + (size_t)(bb * 16 + t) * (66 * 66 * 32);
  const _Float16* hfrm = hprev + (size_t)bb * (66 * 66 * 64);

  // load A (2 frags) + B (4 frags) for x-conv k-block kb
  auto loadX = [&](int kb, Frag16& a0, Frag16& a1, Frag16* bf) {
    const int ky = kb / 3, kx = kb % 3;    // padded coords
    const _Float16* p0 = xfrm + ((size_t)(yy + ky) * 66 + (xx + kx)) * 32 + kOff;
    a0.q[0] = *(const f32x4*)(p0);
    a0.q[1] = *(const f32x4*)(p0 + 16);
    a1.q[0] = *(const f32x4*)(p0 + 512);   // +16 pixels * 32ch
    a1.q[1] = *(const f32x4*)(p0 + 528);
    const _Float16* wb = wxp + ((size_t)((kb * 16 + wv * 4) * 32 + lane)) * 16;
    __builtin_prefetch(wb + 2 * 16 * 32 * 16, 0, 3);
#pragma unroll
    for (int nt = 0; nt < 4; ++nt) {
      bf[nt].q[0] = *(const f32x4*)(wb + nt * 512);
      bf[nt].q[1] = *(const f32x4*)(wb + nt * 512 + 8);
    }
  };

  // load A (2 frags) + B (4 frags) for h-conv k-block kb
  auto loadH = [&](int kb, Frag16& a0, Frag16& a1, Frag16* bf) {
    const int pos = kb >> 1;
    const int cb  = (kb & 1) * 32;
    const int ky = pos / 3, kx = pos % 3;
    const _Float16* p0 =
        hfrm + ((size_t)(yy + ky) * 66 + (xx + kx)) * 64 + cb + kOff;
    a0.q[0] = *(const f32x4*)(p0);
    a0.q[1] = *(const f32x4*)(p0 + 16);
    a1.q[0] = *(const f32x4*)(p0 + 1024);  // +16 pixels * 64ch
    a1.q[1] = *(const f32x4*)(p0 + 1040);
    const _Float16* wb = whp + ((size_t)((kb * 16 + wv * 4) * 32 + lane)) * 16;
    __builtin_prefetch(wb + 2 * 16 * 32 * 16, 0, 3);
#pragma unroll
    for (int nt = 0; nt < 4; ++nt) {
      bf[nt].q[0] = *(const f32x4*)(wb + nt * 512);
      bf[nt].q[1] = *(const f32x4*)(wb + nt * 512 + 8);
    }
  };

  // accumulators [mtile][ntile], init with bias (same N for all 8 rows)
  v8f acc[2][4];
#pragma unroll
  for (int nt = 0; nt < 4; ++nt) {
    float bv = bias[wv * 64 + nt * 16 + nl];
#pragma unroll
    for (int r = 0; r < 8; ++r) { acc[0][nt][r] = bv; acc[1][nt][r] = bv; }
  }

  auto mm8 = [&](Frag16& a0, Frag16& a1, Frag16* bf) {
#pragma unroll
    for (int nt = 0; nt < 4; ++nt) {
      acc[0][nt] = __builtin_amdgcn_wmma_f32_16x16x32_f16(
          false, a0.v, false, bf[nt].v, (short)0, acc[0][nt], false, false);
      acc[1][nt] = __builtin_amdgcn_wmma_f32_16x16x32_f16(
          false, a1.v, false, bf[nt].v, (short)0, acc[1][nt], false, false);
    }
  };

  // ---- software-pipelined K loop: 9 x-blocks then 18 h-blocks ----
  Frag16 ca0, ca1, cbf[4];
  loadX(0, ca0, ca1, cbf);
#pragma unroll
  for (int kb = 0; kb < 9; ++kb) {
    Frag16 na0, na1, nbf[4];
    if (kb < 8) loadX(kb + 1, na0, na1, nbf);
    else        loadH(0, na0, na1, nbf);       // bridge into h section
    mm8(ca0, ca1, cbf);
    ca0 = na0; ca1 = na1;
#pragma unroll
    for (int nt = 0; nt < 4; ++nt) cbf[nt] = nbf[nt];
  }
#pragma unroll
  for (int kb = 0; kb < 18; ++kb) {
    Frag16 na0, na1, nbf[4];
    if (kb < 17) loadH(kb + 1, na0, na1, nbf);
    mm8(ca0, ca1, cbf);
    if (kb < 17) {
      ca0 = na0; ca1 = na1;
#pragma unroll
      for (int nt = 0; nt < 4; ++nt) cbf[nt] = nbf[nt];
    }
  }

  // ---- gates via LDS: zs[gate][pixel 0..31][channel 0..63] ----
  __shared__ float zs[4][32][64];
#pragma unroll
  for (int mt = 0; mt < 2; ++mt)
#pragma unroll
    for (int nt = 0; nt < 4; ++nt)
#pragma unroll
      for (int r = 0; r < 8; ++r) {
        const int mm = mt * 16 + r + ((lane >> 4) << 3);
        zs[wv][mm][nt * 16 + nl] = acc[mt][nt][r];
      }
  __syncthreads();

  // 128 threads x 16 channels: p = tid/4 (0..31), ch = (tid%4)*16 ..
  const int p   = threadIdx.x >> 2;
  const int cb2 = (threadIdx.x & 3) * 16;
  const int pgp = pgBase + p;
  const int b2 = pgp >> 12, y2 = (pgp >> 6) & 63, x2 = pgp & 63;
  const size_t cidx = (size_t)pgp * 64 + cb2;
  const size_t hidx = (((size_t)b2 * 66 + (y2 + 1)) * 66 + (x2 + 1)) * 64 + cb2;
  float* outp =
      out + ((((size_t)b2 * 16 + t) * 64 + y2) * 64 + x2) * 128 + dir * 64 + cb2;
#pragma unroll 4
  for (int j = 0; j < 16; ++j) {
    const float zi = zs[0][p][cb2 + j];
    const float zf = zs[1][p][cb2 + j];
    const float zc = zs[2][p][cb2 + j];
    const float zo = zs[3][p][cb2 + j];
    const float ig = hsig(zi), fg = hsig(zf), og = hsig(zo);
    const float cn = fg * cbuf[cidx + j] + ig * tanhf(zc);
    cbuf[cidx + j] = cn;
    const float h = og * tanhf(cn);
    outp[j] = h;
    hnext[hidx + j] = (_Float16)h;
  }
}

// ---------------------------------------------------------------------------
extern "C" void kernel_launch(void* const* d_in, const int* in_sizes, int n_in,
                              void* d_out, int out_size, void* d_ws,
                              size_t ws_size, hipStream_t stream) {
  (void)in_sizes; (void)n_in; (void)out_size; (void)ws_size;
  const float* x   = (const float*)d_in[0];
  const float* Wxf = (const float*)d_in[1];
  const float* Whf = (const float*)d_in[2];
  const float* bf  = (const float*)d_in[3];
  const float* Wxb = (const float*)d_in[4];
  const float* Whb = (const float*)d_in[5];
  const float* bbk = (const float*)d_in[6];
  float* out = (float*)d_out;

  char* ws = (char*)d_ws;
  _Float16* xpad = (_Float16*)(ws + OFF_XPAD);
  _Float16* wxfp = (_Float16*)(ws + OFF_WXF);
  _Float16* whfp = (_Float16*)(ws + OFF_WHF);
  _Float16* wxbp = (_Float16*)(ws + OFF_WXB);
  _Float16* whbp = (_Float16*)(ws + OFF_WHB);
  float*    cbuf = (float*)(ws + OFF_C);
  _Float16* h0   = (_Float16*)(ws + OFF_H0);
  _Float16* h1   = (_Float16*)(ws + OFF_H1);

  // pack weights into WMMA B-fragment layout (f16)
  pack_w_kernel<<<288, 256, 0, stream>>>(Wxf, wxfp, 288);
  pack_w_kernel<<<576, 256, 0, stream>>>(Whf, whfp, 576);
  pack_w_kernel<<<288, 256, 0, stream>>>(Wxb, wxbp, 288);
  pack_w_kernel<<<576, 256, 0, stream>>>(Whb, whbp, 576);

  // zero padded x (halo), then scatter-convert interior
  zero_f16_kernel<<<8192, 256, 0, stream>>>(xpad, 17842176);
  cvt_pad_kernel<<<8192, 256, 0, stream>>>(x, xpad, 16777216);

  const int nblocks = (8 * 64 * 64) / 32;  // 1024 pixel tiles per step
  for (int dir = 0; dir < 2; ++dir) {
    zero_f32_kernel<<<2048, 256, 0, stream>>>(cbuf, 2097152);
    zero_f16_kernel<<<2048, 256, 0, stream>>>(h0, 2230272);
    zero_f16_kernel<<<2048, 256, 0, stream>>>(h1, 2230272);
    const _Float16* wx = dir ? wxbp : wxfp;
    const _Float16* wh = dir ? whbp : whfp;
    const float*    bs = dir ? bbk  : bf;
    for (int s = 0; s < 16; ++s) {
      const int t = dir ? (15 - s) : s;
      _Float16* hin  = (s & 1) ? h1 : h0;
      _Float16* hout = (s & 1) ? h0 : h1;
      convlstm_step_kernel<<<nblocks, 128, 0, stream>>>(
          xpad, wx, wh, bs, hin, hout, cbuf, out, t, dir);
    }
  }
}
